// PointNet2SpiralsAutoEncoder_82231443849264
// MI455X (gfx1250) — matrix operand order repaired and checked by reading, not compiled, not run on hardware
//
#include <hip/hip_runtime.h>

// ---------------------------------------------------------------------------
// Problem constants (from reference)
// ---------------------------------------------------------------------------
#define BB    16
#define NPTS  5023
#define NP1   128
#define NS1   32
#define RAD1  0.2f
#define NP2   32
#define NS2   64
#define RAD2  0.4f
#define SPK   20

typedef __attribute__((ext_vector_type(16))) _Float16 v16h;
typedef __attribute__((ext_vector_type(8)))  _Float16 v8h;
typedef __attribute__((ext_vector_type(8)))  float    v8f;

// ---------------------------------------------------------------------------
// Farthest point sampling: one block per batch, dist array lives in LDS.
// ---------------------------------------------------------------------------
#define FPS_T 256
__global__ void fps_kernel(const float* __restrict__ xyz, int n, int npoint,
                           int* __restrict__ fidx, float* __restrict__ nx) {
  int b = blockIdx.x;
  const float* X = xyz + (size_t)b * n * 3;
  __shared__ float dist[5024];
  __shared__ float rv[FPS_T];
  __shared__ int   ri[FPS_T];
  __shared__ int   s_far;
  int tid = threadIdx.x;
  for (int i = tid; i < n; i += FPS_T) dist[i] = 1e10f;
  if (tid == 0) s_far = 0;
  __syncthreads();
  for (int it = 0; it < npoint; ++it) {
    int far = s_far;
    float cx = X[far * 3 + 0], cy = X[far * 3 + 1], cz = X[far * 3 + 2];
    if (tid == 0) {
      fidx[b * npoint + it] = far;
      nx[((size_t)b * npoint + it) * 3 + 0] = cx;
      nx[((size_t)b * npoint + it) * 3 + 1] = cy;
      nx[((size_t)b * npoint + it) * 3 + 2] = cz;
    }
    float best = -1.0f; int bi = 0;
    for (int i = tid; i < n; i += FPS_T) {
      float dx = X[i * 3 + 0] - cx;
      float dy = X[i * 3 + 1] - cy;
      float dz = X[i * 3 + 2] - cz;
      float d = dx * dx + dy * dy + dz * dz;
      float od = dist[i];
      d = fminf(od, d);
      dist[i] = d;
      if (d > best) { best = d; bi = i; }
    }
    rv[tid] = best; ri[tid] = bi;
    __syncthreads();
    for (int s = FPS_T / 2; s > 0; s >>= 1) {
      if (tid < s) {
        bool take = (rv[tid + s] > rv[tid]) ||
                    (rv[tid + s] == rv[tid] && ri[tid + s] < ri[tid]);
        if (take) { rv[tid] = rv[tid + s]; ri[tid] = ri[tid + s]; }
      }
      __syncthreads();
    }
    if (tid == 0) s_far = ri[0];
    __syncthreads();
  }
}

// ---------------------------------------------------------------------------
// Ball query: first `nsample` in-radius indices (ascending), padded w/ first.
// ---------------------------------------------------------------------------
__global__ void ball_query_kernel(const float* __restrict__ xyz,
                                  const float* __restrict__ nx,
                                  int n, int npoint, int nsample, float r2,
                                  int* __restrict__ idx, int total) {
  int t = blockIdx.x * blockDim.x + threadIdx.x;
  if (t >= total) return;
  int b = t / npoint;
  const float* X = xyz + (size_t)b * n * 3;
  float cx = nx[(size_t)t * 3 + 0];
  float cy = nx[(size_t)t * 3 + 1];
  float cz = nx[(size_t)t * 3 + 2];
  int* out = idx + (size_t)t * nsample;
  int cnt = 0, first = 0; bool hasf = false;
  for (int i = 0; i < n && cnt < nsample; ++i) {
    float dx = X[i * 3 + 0] - cx;
    float dy = X[i * 3 + 1] - cy;
    float dz = X[i * 3 + 2] - cz;
    float d2 = dx * dx + dy * dy + dz * dz;
    if (d2 <= r2) {
      if (!hasf) { first = i; hasf = true; }
      out[cnt++] = i;
    }
  }
  for (int j = cnt; j < nsample; ++j) out[j] = first;
}

// ---------------------------------------------------------------------------
// f32 -> f16 with row-stride padding: out (R, Cpad), pad = 0
// ---------------------------------------------------------------------------
__global__ void cvt_pad_kernel(const float* __restrict__ in, _Float16* __restrict__ out,
                               int C, int Cpad, int total) {
  int t = blockIdx.x * blockDim.x + threadIdx.x;
  if (t >= total) return;
  int r = t / Cpad, c = t % Cpad;
  out[t] = (c < C) ? (_Float16)in[(size_t)r * C + c] : (_Float16)0.f;
}

// ---------------------------------------------------------------------------
// Batched f16 transpose: in (Z, R, C) -> out (Z, C, Rpad)  (pad region unused)
// ---------------------------------------------------------------------------
__global__ void transpose_h_kernel(const _Float16* __restrict__ in,
                                   _Float16* __restrict__ out,
                                   int R, int C, int Rpad, long inB, long outB,
                                   int total) {
  int z = blockIdx.z;
  int t = blockIdx.x * blockDim.x + threadIdx.x;
  if (t >= total) return;
  int r = t / C, c = t % C;
  out[(size_t)z * outB + (size_t)c * Rpad + r] = in[(size_t)z * inB + t];
}

// ---------------------------------------------------------------------------
// Grouping (gather + center) kernels; f16 outputs, padded row strides
// ---------------------------------------------------------------------------
__global__ void group1_kernel(const float* __restrict__ xyz,
                              const float* __restrict__ nx,
                              const int* __restrict__ idx,
                              _Float16* __restrict__ grp, int total3) {
  int t = blockIdx.x * blockDim.x + threadIdx.x;
  if (t >= total3) return;
  int row = t / 3, c = t % 3;
  int b = row / (NP1 * NS1);
  int s = (row / NS1) % NP1;
  int src = idx[row];
  grp[(size_t)row * 8 + c] = (_Float16)(
      xyz[((size_t)b * NPTS + src) * 3 + c] - nx[((size_t)b * NP1 + s) * 3 + c]);
}

__global__ void group2_kernel(const float* __restrict__ l1x,
                              const _Float16* __restrict__ l1p,
                              const float* __restrict__ nx2,
                              const int* __restrict__ idx2,
                              _Float16* __restrict__ grp, int total) {
  int t = blockIdx.x * blockDim.x + threadIdx.x;
  if (t >= total) return;
  int row = t / 131, c = t % 131;
  int b = row / (NP2 * NS2);
  int s = (row / NS2) % NP2;
  int src = idx2[row];
  float v;
  if (c < 3)
    v = l1x[((size_t)b * NP1 + src) * 3 + c] - nx2[((size_t)b * NP2 + s) * 3 + c];
  else
    v = (float)l1p[((size_t)b * NP1 + src) * 128 + (c - 3)];
  grp[(size_t)row * 136 + c] = (_Float16)v;
}

__global__ void group3_kernel(const float* __restrict__ l2x,
                              const _Float16* __restrict__ l2p,
                              _Float16* __restrict__ grp, int total) {
  int t = blockIdx.x * blockDim.x + threadIdx.x;
  if (t >= total) return;
  int row = t / 259, c = t % 259;  // row = b*32 + kk
  float v;
  if (c < 3) v = l2x[(size_t)row * 3 + c];
  else       v = (float)l2p[(size_t)row * 256 + (c - 3)];
  grp[(size_t)row * 264 + c] = (_Float16)v;
}

// ---------------------------------------------------------------------------
// Max-pool over the sample dimension (f16): in (groups, ns, C) -> out (groups, C)
// ---------------------------------------------------------------------------
__global__ void maxpool_kernel(const _Float16* __restrict__ in,
                               _Float16* __restrict__ out,
                               int groups, int ns, int C) {
  int t = blockIdx.x * blockDim.x + threadIdx.x;
  if (t >= groups * C) return;
  int g = t / C, c = t % C;
  const _Float16* p = in + (size_t)g * ns * C + c;
  float m = (float)p[0];
  for (int k = 1; k < ns; ++k) m = fmaxf(m, (float)p[(size_t)k * C]);
  out[t] = (_Float16)m;
}

// ---------------------------------------------------------------------------
// WMMA GEMM (all-f16 operands, f32 accumulate, NT 16-col tiles per wave):
//   D = act(gamma*(A@B + bias) + beta)
//   A[m][k] = A[m*aRS + k]  (f16, aRS % 8 == 0, 16B-aligned vector loads)
//   GATHER: A[m][k] = A[gidx[m*gK + (k>>gSh)]*aRS + (k & (aRS-1))], aRS = 2^gSh
//   Bmat[k][n] = B[n*bRS + k] (f16, bRS % 8 == 0; weights pre-converted,
//                              activations pre-transposed)
// Main K-loop is branch-free (clamped loads, guarded stores); single guarded
// scalar tail for K % 32 != 0 (never happens in GATHER mode).
// 4 waves / block, one 16-row stripe each. Fragment layouts per ISA 7.12.2.
// ---------------------------------------------------------------------------
template <int NT, bool GATHER, bool DF16>
__global__ __launch_bounds__(128)
void wmma_gemm_t(const _Float16* __restrict__ A, long aBatch, int aRS,
                 const int* __restrict__ gidx, int gK, int gSh,
                 const _Float16* __restrict__ Bh, long bBatch, int bRS,
                 void* __restrict__ Dv, long dBatch, int dRS,
                 int M, int N, int K,
                 const float* __restrict__ bias,
                 const float* __restrict__ gamma,
                 const float* __restrict__ beta, int act) {
  int z = blockIdx.z;
  A  += (size_t)z * aBatch;
  Bh += (size_t)z * bBatch;

  int wave = threadIdx.x >> 5;
  int lane = threadIdx.x & 31;
  int half = lane >> 4;
  int lm   = lane & 15;
  int mt = (blockIdx.y * 4 + wave) * 16;
  if (mt >= M) return;                       // wave-uniform exit
  int nt = blockIdx.x * (16 * NT);

  const v8f z8 = {0.f, 0.f, 0.f, 0.f, 0.f, 0.f, 0.f, 0.f};
  v8f acc[NT];
#pragma unroll
  for (int i = 0; i < NT; ++i) acc[i] = z8;

  int mRow = mt + lm;
  int mC = (mRow < M) ? mRow : (M - 1);      // clamp: loads valid, store guarded
  int nC[NT];
#pragma unroll
  for (int i = 0; i < NT; ++i) {
    int n = nt + 16 * i + lm;
    nC[i] = (n < N) ? n : (N - 1);
  }

  const _Float16* Arow = A + (size_t)mC * aRS;       // non-gather A base
  const int*      Grow = GATHER ? (gidx + mC * gK) : nullptr;
  const _Float16* Brow[NT];
#pragma unroll
  for (int i = 0; i < NT; ++i) Brow[i] = Bh + (size_t)nC[i] * bRS + half * 8;

  const int KMain = K & ~31;
  for (int k0 = 0; k0 < KMain; k0 += 32) {
    v16h a;
    if constexpr (GATHER) {
      int ka = k0 + half * 8;
      int kb = ka + 16;
      int cMask = aRS - 1;                   // aRS is a power of two here
      v8h lo = *(const v8h*)(A + (size_t)Grow[ka >> gSh] * aRS + (ka & cMask));
      v8h hi = *(const v8h*)(A + (size_t)Grow[kb >> gSh] * aRS + (kb & cMask));
      a = __builtin_shufflevector(lo, hi, 0, 1, 2, 3, 4, 5, 6, 7,
                                  8, 9, 10, 11, 12, 13, 14, 15);
    } else {
      v8h lo = *(const v8h*)(Arow + k0 + half * 8);
      v8h hi = *(const v8h*)(Arow + k0 + half * 8 + 16);
      a = __builtin_shufflevector(lo, hi, 0, 1, 2, 3, 4, 5, 6, 7,
                                  8, 9, 10, 11, 12, 13, 14, 15);
    }
#pragma unroll
    for (int i = 0; i < NT; ++i) {
      const v8h* pb = (const v8h*)(Brow[i] + k0);
      v8h b0 = pb[0];
      v8h b1 = pb[2];
      v16h b = __builtin_shufflevector(b0, b1, 0, 1, 2, 3, 4, 5, 6, 7,
                                       8, 9, 10, 11, 12, 13, 14, 15);
      acc[i] = __builtin_amdgcn_wmma_f32_16x16x32_f16(false, a, false, b,
                                                      (short)0, acc[i], false, false);
    }
  }

  if constexpr (!GATHER) {
    if (KMain < K) {                        // single guarded scalar tail
      v16h a;
      int ka = KMain + half * 8;
#pragma unroll
      for (int j = 0; j < 8; ++j) {
        int kA0 = ka + j, kA1 = ka + 16 + j;
        a[j]     = (kA0 < K) ? Arow[kA0] : (_Float16)0.f;
        a[8 + j] = (kA1 < K) ? Arow[kA1] : (_Float16)0.f;
      }
      int kb0 = KMain + half * 8;
#pragma unroll
      for (int i = 0; i < NT; ++i) {
        const _Float16* Bn = Bh + (size_t)nC[i] * bRS;
        v16h b;
#pragma unroll
        for (int t = 0; t < 16; ++t) {
          int kb = kb0 + t + (t >= 8 ? 8 : 0);
          b[t] = (kb < K) ? Bn[kb] : (_Float16)0.f;
        }
        acc[i] = __builtin_amdgcn_wmma_f32_16x16x32_f16(false, a, false, b,
                                                        (short)0, acc[i], false, false);
      }
    }
  }

  // ----- epilogue -----
#pragma unroll
  for (int i = 0; i < NT; ++i) {
    int n = nt + 16 * i + lm;
    if (n < N) {
      float bb = bias  ? bias[n]  : 0.f;
      float gg = gamma ? gamma[n] : 1.f;
      float ee = beta  ? beta[n]  : 0.f;
#pragma unroll
      for (int r = 0; r < 8; ++r) {
        int m = mt + r + 8 * half;          // C/D: VGPR r -> row r (+8 for half 1)
        if (m < M) {
          float v = acc[i][r];
          v = gg * (v + bb) + ee;
          if (act == 1)      v = fmaxf(v, 0.f);
          else if (act == 2) v = (v > 0.f) ? v : (__expf(v) - 1.f);
          if constexpr (DF16)
            ((_Float16*)Dv)[(size_t)z * dBatch + (size_t)m * dRS + n] = (_Float16)v;
          else
            ((float*)Dv)[(size_t)z * dBatch + (size_t)m * dRS + n] = v;
        }
      }
    }
  }
}

template <int NT, bool GATHER, bool DF16>
static void gemm_launch(hipStream_t st, const _Float16* A, long aB, int aRS,
                        const int* gi, int gSh,
                        const _Float16* Bh, long bB, int bRS,
                        void* D, long dB, int dRS, int M, int N, int K,
                        const float* bias, const float* gam, const float* bet,
                        int act, int Z) {
  dim3 g((N + 16 * NT - 1) / (16 * NT), (M + 63) / 64, (unsigned)Z);
  wmma_gemm_t<NT, GATHER, DF16><<<g, dim3(128), 0, st>>>(
      A, aB, aRS, gi, SPK, gSh, Bh, bB, bRS, D, dB, dRS, M, N, K,
      bias, gam, bet, act);
}

// ---------------------------------------------------------------------------
// Host-side orchestration
// ---------------------------------------------------------------------------
extern "C" void kernel_launch(void* const* d_in, const int* in_sizes, int n_in,
                              void* d_out, int out_size, void* d_ws, size_t ws_size,
                              hipStream_t stream) {
  // ---- unpack inputs (flattened pytree order) ----
  const float* x = (const float*)d_in[0];
  const float *Wm[3][3], *Gm[3][3], *Bem[3][3];
  int p = 1;
  for (int s = 0; s < 3; ++s)
    for (int l = 0; l < 3; ++l) {
      Wm[s][l]  = (const float*)d_in[p++];
      Gm[s][l]  = (const float*)d_in[p++];
      Bem[s][l] = (const float*)d_in[p++];
    }
  const float* W1  = (const float*)d_in[p++]; const float* b1  = (const float*)d_in[p++];
  const float* g1  = (const float*)d_in[p++]; const float* be1 = (const float*)d_in[p++];
  const float* W2  = (const float*)d_in[p++]; const float* b2  = (const float*)d_in[p++];
  const float* g2  = (const float*)d_in[p++]; const float* be2 = (const float*)d_in[p++];
  const float* W3  = (const float*)d_in[p++]; const float* b3  = (const float*)d_in[p++];
  const float* Wl  = (const float*)d_in[p++]; const float* bl  = (const float*)d_in[p++];
  const float* Wc0 = (const float*)d_in[p++]; const float* bc0 = (const float*)d_in[p++];
  const float* Wc1 = (const float*)d_in[p++]; const float* bc1 = (const float*)d_in[p++];
  const float* Wc2 = (const float*)d_in[p++]; const float* bc2 = (const float*)d_in[p++];
  const float* U0  = (const float*)d_in[p++];   // (5024,1257)
  const float* U1  = (const float*)d_in[p++];   // (1257,315)
  const float* U2  = (const float*)d_in[p++];   // (315,80)
  const int*   S0  = (const int*)d_in[p++];     // (5024,20)
  const int*   S1  = (const int*)d_in[p++];     // (1257,20)
  const int*   S2  = (const int*)d_in[p++];     // (315,20)
  (void)in_sizes; (void)n_in; (void)out_size; (void)ws_size;

  // ---- carve workspace ----
  char* w = (char*)d_ws;
  size_t off = 0;
  auto carve = [&](size_t bytes) -> void* {
    void* ptr = (void*)(w + off);
    off = (off + bytes + 255) & ~(size_t)255;
    return ptr;
  };
  int*      fidx1 = (int*)carve((size_t)BB * NP1 * 4);
  float*    nx1   = (float*)carve((size_t)BB * NP1 * 3 * 4);       // == l1x
  int*      idx1  = (int*)carve((size_t)BB * NP1 * NS1 * 4);
  int*      fidx2 = (int*)carve((size_t)BB * NP2 * 4);
  float*    nx2   = (float*)carve((size_t)BB * NP2 * 3 * 4);       // == l2x
  int*      idx2  = (int*)carve((size_t)BB * NP2 * NS2 * 4);
  _Float16* grp   = (_Float16*)carve((size_t)BB * NP2 * NS2 * 136 * 2); // shared
  _Float16* l1p   = (_Float16*)carve((size_t)BB * NP1 * 128 * 2);
  _Float16* l2p   = (_Float16*)carve((size_t)BB * NP2 * 256 * 2);
  _Float16* l3p   = (_Float16*)carve((size_t)BB * 1024 * 2);
  _Float16* f1    = (_Float16*)carve((size_t)BB * 512 * 2);
  _Float16* f2    = (_Float16*)carve((size_t)BB * 256 * 2);
  _Float16* zz    = (_Float16*)carve((size_t)BB * 128 * 2);
  _Float16* bufA  = (_Float16*)carve((size_t)8388608 * 2);
  _Float16* bufB  = (_Float16*)carve((size_t)8388608 * 2);
  _Float16* hT    = (_Float16*)carve((size_t)BB * 32 * 1264 * 2);  // largest h^T

  auto cdiv = [](int a, int b) { return (a + b - 1) / b; };

  // ---- convert all weights / U matrices to padded f16 (once per launch) ----
  const float* wsrc[19] = {Wm[0][0], Wm[0][1], Wm[0][2], Wm[1][0], Wm[1][1], Wm[1][2],
                           Wm[2][0], Wm[2][1], Wm[2][2], W1, W2, W3, Wl,
                           Wc0, Wc1, Wc2, U2, U1, U0};
  const int wR[19] = {64, 64, 128, 128, 128, 256, 256, 512, 1024,
                      512, 256, 128, 10240, 64, 32, 3, 315, 1257, 5024};
  const int wC[19] = {3, 64, 64, 131, 128, 128, 259, 256, 512,
                      1024, 512, 256, 128, 2560, 1280, 640, 80, 315, 1257};
  const int wP[19] = {8, 64, 64, 136, 128, 128, 264, 256, 512,
                      1024, 512, 256, 128, 2560, 1280, 640, 80, 320, 1264};
  _Float16* wh[19];
  for (int i = 0; i < 19; ++i) {
    int total = wR[i] * wP[i];
    wh[i] = (_Float16*)carve((size_t)total * 2);
    cvt_pad_kernel<<<cdiv(total, 256), 256, 0, stream>>>(wsrc[i], wh[i],
                                                         wC[i], wP[i], total);
  }
  _Float16 *U2h = wh[16], *U1h = wh[17], *U0h = wh[18];

  // ======================= Encoder: SA1 =======================
  fps_kernel<<<BB, FPS_T, 0, stream>>>(x, NPTS, NP1, fidx1, nx1);
  {
    int total = BB * NP1;
    ball_query_kernel<<<cdiv(total, 128), 128, 0, stream>>>(
        x, nx1, NPTS, NP1, NS1, RAD1 * RAD1, idx1, total);
  }
  {
    int t3 = BB * NP1 * NS1 * 3;
    group1_kernel<<<cdiv(t3, 256), 256, 0, stream>>>(x, nx1, idx1, grp, t3);
  }
  const int M1 = BB * NP1 * NS1;  // 65536
  gemm_launch<4, false, true>(stream, grp,  0, 8,   nullptr, 0, wh[0], 0, 8,   bufA, 0, 64,  M1, 64,  3,   nullptr, Gm[0][0], Bem[0][0], 1, 1);
  gemm_launch<4, false, true>(stream, bufA, 0, 64,  nullptr, 0, wh[1], 0, 64,  bufB, 0, 64,  M1, 64,  64,  nullptr, Gm[0][1], Bem[0][1], 1, 1);
  gemm_launch<4, false, true>(stream, bufB, 0, 64,  nullptr, 0, wh[2], 0, 64,  bufA, 0, 128, M1, 128, 64,  nullptr, Gm[0][2], Bem[0][2], 1, 1);
  maxpool_kernel<<<cdiv(BB * NP1 * 128, 256), 256, 0, stream>>>(bufA, l1p, BB * NP1, NS1, 128);

  // ======================= Encoder: SA2 =======================
  fps_kernel<<<BB, FPS_T, 0, stream>>>(nx1, NP1, NP2, fidx2, nx2);
  {
    int total = BB * NP2;
    ball_query_kernel<<<cdiv(total, 128), 128, 0, stream>>>(
        nx1, nx2, NP1, NP2, NS2, RAD2 * RAD2, idx2, total);
  }
  {
    int tt = BB * NP2 * NS2 * 131;
    group2_kernel<<<cdiv(tt, 256), 256, 0, stream>>>(nx1, l1p, nx2, idx2, grp, tt);
  }
  const int M2 = BB * NP2 * NS2;  // 32768
  gemm_launch<4, false, true>(stream, grp,  0, 136, nullptr, 0, wh[3], 0, 136, bufA, 0, 128, M2, 128, 131, nullptr, Gm[1][0], Bem[1][0], 1, 1);
  gemm_launch<4, false, true>(stream, bufA, 0, 128, nullptr, 0, wh[4], 0, 128, bufB, 0, 128, M2, 128, 128, nullptr, Gm[1][1], Bem[1][1], 1, 1);
  gemm_launch<4, false, true>(stream, bufB, 0, 128, nullptr, 0, wh[5], 0, 128, bufA, 0, 256, M2, 256, 128, nullptr, Gm[1][2], Bem[1][2], 1, 1);
  maxpool_kernel<<<cdiv(BB * NP2 * 256, 256), 256, 0, stream>>>(bufA, l2p, BB * NP2, NS2, 256);

  // ======================= Encoder: SA3 (global) =======================
  {
    int tt = BB * 32 * 259;
    group3_kernel<<<cdiv(tt, 256), 256, 0, stream>>>(nx2, l2p, grp, tt);
  }
  const int M3 = BB * 32;  // 512
  gemm_launch<4, false, true>(stream, grp,  0, 264, nullptr, 0, wh[6], 0, 264, bufA, 0, 256,  M3, 256,  259, nullptr, Gm[2][0], Bem[2][0], 1, 1);
  gemm_launch<4, false, true>(stream, bufA, 0, 256, nullptr, 0, wh[7], 0, 256, bufB, 0, 512,  M3, 512,  256, nullptr, Gm[2][1], Bem[2][1], 1, 1);
  gemm_launch<4, false, true>(stream, bufB, 0, 512, nullptr, 0, wh[8], 0, 512, bufA, 0, 1024, M3, 1024, 512, nullptr, Gm[2][2], Bem[2][2], 1, 1);
  maxpool_kernel<<<cdiv(BB * 1024, 256), 256, 0, stream>>>(bufA, l3p, BB, 32, 1024);

  // ======================= FC bottleneck =======================
  gemm_launch<4, false, true>(stream, l3p, 0, 1024, nullptr, 0, wh[9],  0, 1024, f1, 0, 512, BB, 512, 1024, b1, g1, be1, 1, 1);
  gemm_launch<4, false, true>(stream, f1,  0, 512,  nullptr, 0, wh[10], 0, 512,  f2, 0, 256, BB, 256, 512,  b2, g2, be2, 1, 1);
  gemm_launch<4, false, true>(stream, f2,  0, 256,  nullptr, 0, wh[11], 0, 256,  zz, 0, 128, BB, 128, 256,  b3, nullptr, nullptr, 0, 1);

  // ======================= Decoder =======================
  // dec_fc: z(16,128) @ Wl^T -> h0 (16,80,128) f16 in bufA
  gemm_launch<4, false, true>(stream, zz, 0, 128, nullptr, 0, wh[12], 0, 128, bufA, 0, 10240, BB, 10240, 128, bl, nullptr, nullptr, 0, 1);

  // stage 0: hT = h0^T (128,80); U2 (315,80) x h0 -> (B,315,128); conv K=2560 -> 64, ELU
  transpose_h_kernel<<<dim3(cdiv(80 * 128, 256), 1, BB), 256, 0, stream>>>(
      bufA, hT, 80, 128, 80, 10240L, 10240L, 80 * 128);
  gemm_launch<4, false, true>(stream, U2h, 0, 80, nullptr, 0, hT, 10240L, 80,
      bufB, 315L * 128, 128, 315, 128, 80, nullptr, nullptr, nullptr, 0, BB);
  gemm_launch<4, true, true>(stream, bufB, 315L * 128, 128, S2, 7, wh[13], 0, 2560,
      bufA, 315L * 64, 64, 315, 64, 2560, bc0, nullptr, nullptr, 2, BB);

  // stage 1: hT = hc0^T (64,320); U1 (1257,315) x h -> (B,1257,64); conv K=1280 -> 32, ELU
  transpose_h_kernel<<<dim3(cdiv(315 * 64, 256), 1, BB), 256, 0, stream>>>(
      bufA, hT, 315, 64, 320, 315L * 64, 64L * 320, 315 * 64);
  gemm_launch<4, false, true>(stream, U1h, 0, 320, nullptr, 0, hT, 64L * 320, 320,
      bufB, 1257L * 64, 64, 1257, 64, 315, nullptr, nullptr, nullptr, 0, BB);
  gemm_launch<2, true, true>(stream, bufB, 1257L * 64, 64, S1, 6, wh[14], 0, 1280,
      bufA, 1257L * 32, 32, 1257, 32, 1280, bc1, nullptr, nullptr, 2, BB);

  // stage 2: hT = hc1^T (32,1264); U0 (5024,1257) x h -> (B,5024,32); conv K=640 -> 3 -> d_out
  transpose_h_kernel<<<dim3(cdiv(1257 * 32, 256), 1, BB), 256, 0, stream>>>(
      bufA, hT, 1257, 32, 1264, 1257L * 32, 32L * 1264, 1257 * 32);
  gemm_launch<2, false, true>(stream, U0h, 0, 1264, nullptr, 0, hT, 32L * 1264, 1264,
      bufB, 5024L * 32, 32, 5024, 32, 1257, nullptr, nullptr, nullptr, 0, BB);
  gemm_launch<1, true, false>(stream, bufB, 5024L * 32, 32, S0, 5, wh[15], 0, 640,
      d_out, 5024L * 3, 3, 5024, 3, 640, bc2, nullptr, nullptr, 0, BB);
}
